// KGAT_32341103739255
// MI455X (gfx1250) — matrix operand order repaired and checked by reading, not compiled, not run on hardware
//
#include <hip/hip_runtime.h>

typedef __attribute__((ext_vector_type(16))) _Float16 v16h;
typedef __attribute__((ext_vector_type(8)))  float    v8f;
typedef __attribute__((ext_vector_type(4)))  float    v4f;

#define N_ENT 100000
#define N_REL 12
#define DIM   64
#define NE    1000000
#define NT    (N_ENT / 16)   // 6250 row tiles

static __device__ __forceinline__ float lrelu(float x) { return x > 0.f ? x : 0.01f * x; }

// ---------------------------------------------------------------------------
// Packing kernels: lay out f16 tiles exactly as the WMMA VGPR layouts expect.
// A 16x32 f16 (ISA 7.12.2): lane L (M = L&15), half h:
//   K = (L>=16 ? 8 : 0) + (h<8 ? h : h+8)
// B 32x16 f16: lane L (N = L&15), half h: K = (L>=16 ? 8 : 0) + (h<8 ? h : h+8)
// ---------------------------------------------------------------------------
__global__ void k_pack_entityA(const float* __restrict__ ent, _Float16* __restrict__ apack) {
    int gid = blockIdx.x * blockDim.x + threadIdx.x;
    if (gid >= N_ENT * DIM) return;
    int h = gid & 15, L = (gid >> 4) & 31, s = (gid >> 9) & 1, t = gid >> 10;
    int M = L & 15;
    int K = ((L >= 16) ? 8 : 0) + (h < 8 ? h : h + 8);
    apack[gid] = (_Float16)ent[(size_t)(16 * t + M) * DIM + 32 * s + K];
}

__global__ void k_pack_Wr(const float* __restrict__ Wr, _Float16* __restrict__ bp) {
    int gid = blockIdx.x * blockDim.x + threadIdx.x;
    if (gid >= N_REL * 2 * 4 * 512) return;
    int h = gid & 15, L = (gid >> 4) & 31, nt = (gid >> 9) & 3, s = (gid >> 11) & 1, k = gid >> 12;
    int d = 32 * s + (L < 16 ? 0 : 8) + (h < 8 ? h : h + 8);
    int n = 16 * nt + (L & 15);
    bp[gid] = (_Float16)Wr[((size_t)k * 64 + d) * 64 + n];
}

// W shape [ncols, 64] row-major; B(d, n) = W[n*64 + d]   (x @ W.T)
__global__ void k_pack_lin(const float* __restrict__ W, _Float16* __restrict__ bp, int nnt) {
    int gid = blockIdx.x * blockDim.x + threadIdx.x;
    if (gid >= 2 * nnt * 512) return;
    int h = gid & 15, L = (gid >> 4) & 31;
    int rest = gid >> 9;
    int nt = rest % nnt, s = rest / nnt;
    int d = 32 * s + (L < 16 ? 0 : 8) + (h < 8 ? h : h + 8);
    int n = 16 * nt + (L & 15);
    bp[gid] = (_Float16)W[n * 64 + d];
}

// ---------------------------------------------------------------------------
// proj[n][k][r] = sum_d ent[n,d] * W_r[k,d,r]  via 8 WMMAs per (tile, rel)
// ---------------------------------------------------------------------------
__global__ void k_proj(const v16h* __restrict__ apack, const v16h* __restrict__ bpWr,
                       _Float16* __restrict__ proj) {
    int t = blockIdx.x, k = blockIdx.y, L = threadIdx.x;
    v8f zero = {0.f,0.f,0.f,0.f,0.f,0.f,0.f,0.f};
    v8f c[4];
#pragma unroll
    for (int i = 0; i < 4; ++i) c[i] = zero;
#pragma unroll
    for (int s = 0; s < 2; ++s) {
        v16h a = apack[(size_t)(t * 2 + s) * 32 + L];
#pragma unroll
        for (int nt = 0; nt < 4; ++nt) {
            v16h b = bpWr[(size_t)(((k * 2 + s) * 4) + nt) * 32 + L];
            c[nt] = __builtin_amdgcn_wmma_f32_16x16x32_f16(false, a, false, b,
                                                           (short)0, c[nt], false, false);
        }
    }
    int col = L & 15, Mh = L >> 4;
#pragma unroll
    for (int nt = 0; nt < 4; ++nt)
#pragma unroll
        for (int v = 0; v < 8; ++v) {
            int node = 16 * t + 8 * Mh + v;
            proj[((size_t)node * N_REL + k) * 64 + 16 * nt + col] = (_Float16)c[nt][v];
        }
}

// ---------------------------------------------------------------------------
// Attention
// ---------------------------------------------------------------------------
__global__ void k_init_seg(unsigned* __restrict__ segmax, float* __restrict__ den) {
    int i = blockIdx.x * blockDim.x + threadIdx.x;
    if (i < N_ENT) { segmax[i] = 0xFF800000u; den[i] = 0.f; }  // -inf, 0
}

__global__ void k_edge_logits(const _Float16* __restrict__ proj, const float* __restrict__ rel_emb,
                              const int* __restrict__ src, const int* __restrict__ dst,
                              const int* __restrict__ etype,
                              float* __restrict__ logits, unsigned* __restrict__ segmax) {
    int e = blockIdx.x * blockDim.x + threadIdx.x;
    if (e >= NE) return;
    int sI = src[e], dI = dst[e], k = etype[e];
    const _Float16* tail = proj + ((size_t)sI * N_REL + k) * 64;
    const _Float16* head = proj + ((size_t)dI * N_REL + k) * 64;
    const float* rel = rel_emb + k * 64;
    float acc = 0.f;
#pragma unroll 8
    for (int i = 0; i < 64; ++i) {
        float hv = (float)head[i] + rel[i];
        acc += (float)tail[i] * tanhf(hv);
    }
    logits[e] = acc;
    // float atomic max via monotone int/uint mapping (init = -inf bits)
    if (acc >= 0.f) atomicMax((int*)segmax + dI, __float_as_int(acc));
    else            atomicMin(segmax + dI, __float_as_uint(acc));
}

__global__ void k_edge_exp(const float* __restrict__ logits, const int* __restrict__ dst,
                           const unsigned* __restrict__ segmax,
                           float* __restrict__ att, float* __restrict__ den) {
    int e = blockIdx.x * blockDim.x + threadIdx.x;
    if (e >= NE) return;
    int dI = dst[e];
    float m = __uint_as_float(segmax[dI]);
    float ex = expf(logits[e] - m);
    att[e] = ex;
    atomicAdd(den + dI, ex);
}

__global__ void k_norm_att(float* __restrict__ att, const float* __restrict__ den,
                           const int* __restrict__ dst) {
    int e = blockIdx.x * blockDim.x + threadIdx.x;
    if (e < NE) att[e] = att[e] / den[dst[e]];
}

// ---------------------------------------------------------------------------
// Graph conv: h_n[dst] += x[src] * att   (8 threads per edge, L2-resident atomics)
// ---------------------------------------------------------------------------
__global__ void k_zero(float* __restrict__ p, int n) {
    int i = blockIdx.x * blockDim.x + threadIdx.x;
    if (i < n) p[i] = 0.f;
}

__global__ void k_scatter(const float* __restrict__ x, const float* __restrict__ att,
                          const int* __restrict__ src, const int* __restrict__ dst,
                          float* __restrict__ hn) {
    int tid = blockIdx.x * blockDim.x + threadIdx.x;
    int e = tid >> 3, part = tid & 7;
    if (e >= NE) return;
    float a = att[e];
    const float* xs = x + (size_t)src[e] * DIM + part * 8;
    float* hd = hn + (size_t)dst[e] * DIM + part * 8;
#pragma unroll
    for (int i = 0; i < 8; ++i) atomicAdd(hd + i, xs[i] * a);
}

// ---------------------------------------------------------------------------
// Fused bi-interaction layer:
//   y = lrelu((x+h) @ W1.T + b1) + lrelu((x*h) @ W2.T + b2)
// One wave per 16-row tile; A tiles built in-register from f32 x/h.
// ---------------------------------------------------------------------------
template <int NNT>
__global__ void k_layer(const float* __restrict__ x, const float* __restrict__ hn,
                        const v16h* __restrict__ bp1, const float* __restrict__ bias1,
                        const v16h* __restrict__ bp2, const float* __restrict__ bias2,
                        float* __restrict__ out, int colBase, float* __restrict__ xnext) {
    int t = blockIdx.x, L = threadIdx.x;
    v8f zero = {0.f,0.f,0.f,0.f,0.f,0.f,0.f,0.f};
    v8f c1[NNT], c2[NNT];
#pragma unroll
    for (int i = 0; i < NNT; ++i) { c1[i] = zero; c2[i] = zero; }

    int M = L & 15;
    const float* xr = x  + (size_t)(16 * t + M) * DIM;
    const float* hr = hn + (size_t)(16 * t + M) * DIM;
#pragma unroll
    for (int s = 0; s < 2; ++s) {
        int base = 32 * s + ((L >= 16) ? 8 : 0);
        v4f xa = *(const v4f*)(xr + base);      v4f ha = *(const v4f*)(hr + base);
        v4f xb = *(const v4f*)(xr + base + 4);  v4f hb = *(const v4f*)(hr + base + 4);
        v4f xc = *(const v4f*)(xr + base + 16); v4f hc = *(const v4f*)(hr + base + 16);
        v4f xd = *(const v4f*)(xr + base + 20); v4f hd = *(const v4f*)(hr + base + 20);
        v16h a1, a2;
#pragma unroll
        for (int i = 0; i < 4; ++i) {
            a1[i]      = (_Float16)(xa[i] + ha[i]);
            a1[4 + i]  = (_Float16)(xb[i] + hb[i]);
            a1[8 + i]  = (_Float16)(xc[i] + hc[i]);
            a1[12 + i] = (_Float16)(xd[i] + hd[i]);
            a2[i]      = (_Float16)(xa[i] * ha[i]);
            a2[4 + i]  = (_Float16)(xb[i] * hb[i]);
            a2[8 + i]  = (_Float16)(xc[i] * hc[i]);
            a2[12 + i] = (_Float16)(xd[i] * hd[i]);
        }
#pragma unroll
        for (int nt = 0; nt < NNT; ++nt) {
            v16h b1 = bp1[(size_t)(s * NNT + nt) * 32 + L];
            c1[nt] = __builtin_amdgcn_wmma_f32_16x16x32_f16(false, a1, false, b1,
                                                            (short)0, c1[nt], false, false);
            v16h b2 = bp2[(size_t)(s * NNT + nt) * 32 + L];
            c2[nt] = __builtin_amdgcn_wmma_f32_16x16x32_f16(false, a2, false, b2,
                                                            (short)0, c2[nt], false, false);
        }
    }
    int col = L & 15, Mh = L >> 4;
#pragma unroll
    for (int nt = 0; nt < NNT; ++nt) {
        float bb1 = bias1[16 * nt + col], bb2 = bias2[16 * nt + col];
#pragma unroll
        for (int v = 0; v < 8; ++v) {
            int node = 16 * t + 8 * Mh + v;
            float y = lrelu(c1[nt][v] + bb1) + lrelu(c2[nt][v] + bb2);
            out[(size_t)node * 160 + colBase + 16 * nt + col] = y;
            if (xnext) xnext[(size_t)node * DIM + 16 * nt + col] = y;
        }
    }
}

__global__ void k_copy_x0(const float* __restrict__ ent, float* __restrict__ out) {
    int i = blockIdx.x * blockDim.x + threadIdx.x;
    if (i >= N_ENT * DIM) return;
    int n = i >> 6, c = i & 63;
    out[(size_t)n * 160 + c] = ent[i];
}

// ---------------------------------------------------------------------------
extern "C" void kernel_launch(void* const* d_in, const int* in_sizes, int n_in,
                              void* d_out, int out_size, void* d_ws, size_t ws_size,
                              hipStream_t stream) {
    (void)in_sizes; (void)n_in; (void)out_size; (void)ws_size;
    const float* ent   = (const float*)d_in[0];
    const float* rel   = (const float*)d_in[1];
    const float* Wr    = (const float*)d_in[2];
    const float* W1_0w = (const float*)d_in[3];
    const float* W1_0b = (const float*)d_in[4];
    const float* W2_0w = (const float*)d_in[5];
    const float* W2_0b = (const float*)d_in[6];
    const float* W1_1w = (const float*)d_in[7];
    const float* W1_1b = (const float*)d_in[8];
    const float* W2_1w = (const float*)d_in[9];
    const float* W2_1b = (const float*)d_in[10];
    const int* src = (const int*)d_in[11];
    const int* dst = (const int*)d_in[12];
    const int* et  = (const int*)d_in[13];
    float* out = (float*)d_out;

    char* ws = (char*)d_ws;
    size_t off = 0;
    auto alloc = [&](size_t bytes) -> char* {
        char* p = ws + off;
        off = (off + bytes + 255) & ~(size_t)255;
        return p;
    };
    _Float16* proj   = (_Float16*)alloc((size_t)N_ENT * N_REL * 64 * 2);  // 153.6 MB
    _Float16* apack  = (_Float16*)alloc((size_t)N_ENT * 64 * 2);          // 12.8 MB
    _Float16* bpWr   = (_Float16*)alloc((size_t)N_REL * 2 * 4 * 512 * 2);
    _Float16* bp10   = (_Float16*)alloc(2 * 4 * 512 * 2);
    _Float16* bp20   = (_Float16*)alloc(2 * 4 * 512 * 2);
    _Float16* bp11   = (_Float16*)alloc(2 * 2 * 512 * 2);
    _Float16* bp21   = (_Float16*)alloc(2 * 2 * 512 * 2);
    float*    logits = (float*)alloc((size_t)NE * 4);
    unsigned* segmax = (unsigned*)alloc((size_t)N_ENT * 4);
    float*    den    = (float*)alloc((size_t)N_ENT * 4);
    float*    att    = (float*)alloc((size_t)NE * 4);
    float*    hn     = (float*)alloc((size_t)N_ENT * DIM * 4);
    float*    x1     = (float*)alloc((size_t)N_ENT * DIM * 4);

    const int B = 256;
    // Pack operands into WMMA layouts
    k_pack_entityA<<<(N_ENT * DIM + B - 1) / B, B, 0, stream>>>(ent, apack);
    k_pack_Wr<<<(N_REL * 4096 + B - 1) / B, B, 0, stream>>>(Wr, bpWr);
    k_pack_lin<<<(4096 + B - 1) / B, B, 0, stream>>>(W1_0w, bp10, 4);
    k_pack_lin<<<(4096 + B - 1) / B, B, 0, stream>>>(W2_0w, bp20, 4);
    k_pack_lin<<<(2048 + B - 1) / B, B, 0, stream>>>(W1_1w, bp11, 2);
    k_pack_lin<<<(2048 + B - 1) / B, B, 0, stream>>>(W2_1w, bp21, 2);
    k_init_seg<<<(N_ENT + B - 1) / B, B, 0, stream>>>(segmax, den);

    // Batched relation GEMM (WMMA)
    k_proj<<<dim3(NT, N_REL), 32, 0, stream>>>((const v16h*)apack, (const v16h*)bpWr, proj);

    // Attention: logits, segment-max, exp/segment-sum, normalize
    k_edge_logits<<<(NE + B - 1) / B, B, 0, stream>>>(proj, rel, src, dst, et, logits, segmax);
    k_edge_exp<<<(NE + B - 1) / B, B, 0, stream>>>(logits, dst, segmax, att, den);
    k_norm_att<<<(NE + B - 1) / B, B, 0, stream>>>(att, den, dst);

    // Layer 1: scatter + fused bi-interaction GEMM (WMMA)
    k_zero<<<(N_ENT * DIM + B - 1) / B, B, 0, stream>>>(hn, N_ENT * DIM);
    k_scatter<<<(NE * 8 + B - 1) / B, B, 0, stream>>>(ent, att, src, dst, hn);
    k_layer<4><<<NT, 32, 0, stream>>>(ent, hn, (const v16h*)bp10, W1_0b,
                                      (const v16h*)bp20, W2_0b, out, 64, x1);

    // Layer 2
    k_zero<<<(N_ENT * DIM + B - 1) / B, B, 0, stream>>>(hn, N_ENT * DIM);
    k_scatter<<<(NE * 8 + B - 1) / B, B, 0, stream>>>(x1, att, src, dst, hn);
    k_layer<2><<<NT, 32, 0, stream>>>(x1, hn, (const v16h*)bp11, W1_1b,
                                      (const v16h*)bp21, W2_1b, out, 128, nullptr);

    // x0 passthrough columns
    k_copy_x0<<<(N_ENT * DIM + B - 1) / B, B, 0, stream>>>(ent, out);
}
